// GCN_21474836480022
// MI455X (gfx1250) — matrix operand order repaired and checked by reading, not compiled, not run on hardware
//
#include <hip/hip_runtime.h>
#include <hip/hip_bf16.h>

typedef __attribute__((ext_vector_type(16))) __bf16 v16bf;
typedef __attribute__((ext_vector_type(8)))  float  v8f;

#define KDIM 128

// ---------------- elementwise / prep kernels ----------------

__global__ void cvt_bf16_kernel(const float* __restrict__ x, __bf16* __restrict__ y, int n) {
    int i = blockIdx.x * blockDim.x + threadIdx.x;
    if (i < n) y[i] = (__bf16)x[i];
}

// Pack row-major W[128 x ncols] (f32) into per-lane WMMA B fragments (bf16):
// layout [nt][s][lane][j] with lane<16 -> K = s*32 + j, col = nt*16 + lane
//                         lane>=16 -> K = s*32 + 16 + j, col = nt*16 + (lane-16)
__global__ void pack_w_kernel(const float* __restrict__ W, __bf16* __restrict__ Wp, int ncols) {
    int idx = blockIdx.x * blockDim.x + threadIdx.x;
    int total = (ncols >> 4) * 4 * 32 * 16;
    if (idx >= total) return;
    int j    = idx & 15;
    int lane = (idx >> 4) & 31;
    int s    = (idx >> 9) & 3;
    int nt   = idx >> 11;
    int khalf = (lane >> 4) << 4;          // 0 or 16
    int k    = s * 32 + khalf + j;
    int col  = (nt << 4) + (lane & 15);
    Wp[idx] = (__bf16)W[k * ncols + col];
}

__global__ void degree_kernel(const int* __restrict__ dst, float* __restrict__ deg, int ne) {
    int e = blockIdx.x * blockDim.x + threadIdx.x;
    if (e < ne) atomicAdd(&deg[dst[e]], 1.0f);
}

__global__ void deg_inv_kernel(float* __restrict__ deg, int n) {
    int i = blockIdx.x * blockDim.x + threadIdx.x;
    if (i < n) deg[i] = 1.0f / fmaxf(deg[i], 1.0f);
}

__global__ void relu_norm_kernel(const float* __restrict__ agg, const float* __restrict__ dinv,
                                 const float* __restrict__ b, __bf16* __restrict__ h, int n) {
    int i = blockIdx.x * blockDim.x + threadIdx.x;
    if (i >= n) return;
    int node = i >> 7, f = i & 127;
    float v = agg[i] * dinv[node] + b[f];
    h[i] = (__bf16)fmaxf(v, 0.0f);
}

__global__ void finalize_kernel(float* __restrict__ out, const float* __restrict__ dinv,
                                const float* __restrict__ b, int n) {
    int i = blockIdx.x * blockDim.x + threadIdx.x;
    if (i >= n) return;
    int node = i >> 6, f = i & 63;
    out[i] = out[i] * dinv[node] + b[f];
}

// ---------------- edge aggregation (atomics hit L2-resident buffers) ----------------

__global__ void scatter_add_kernel(const float* __restrict__ M, const int* __restrict__ src,
                                   const int* __restrict__ dst, float* __restrict__ Agg,
                                   int ne, int shift, int F) {
    int idx = blockIdx.x * blockDim.x + threadIdx.x;
    int e = idx >> shift;
    if (e >= ne) return;
    int c = (idx & ((1 << shift) - 1)) << 2;      // 4 floats per thread
    const float4 v = *(const float4*)(M + src[e] * F + c);
    float* p = Agg + dst[e] * F + c;
    atomicAdd(p + 0, v.x);
    atomicAdd(p + 1, v.y);
    atomicAdd(p + 2, v.z);
    atomicAdd(p + 3, v.w);
}

// ---------------- WMMA GEMM: C[rows x ncols] = A[rows x 128] @ W[128 x ncols] ----------------
// blockDim = (ncols/16) waves * 32; one 16-row tile per block, one 16-col tile per wave.
// A tile staged through LDS (padded stride 136 bf16 to spread banks).

__global__ void __launch_bounds__(256)
gemm_bf16_wmma_kernel(const __bf16* __restrict__ A, const __bf16* __restrict__ Bp,
                      float* __restrict__ C, int ncols) {
    __shared__ __bf16 As[16 * 136];
    const int row0 = blockIdx.x << 4;
    const int tid  = threadIdx.x;

    // Cooperative stage of 16 x 128 bf16 A-tile (256 x 16-byte chunks)
    for (int i = tid; i < 256; i += blockDim.x) {
        int r = i >> 4, c = i & 15;
        *(uint4*)(&As[r * 136 + c * 8]) = *(const uint4*)(A + (row0 + r) * KDIM + c * 8);
    }
    __syncthreads();

    const int lane  = tid & 31;
    const int nt    = tid >> 5;                // n-tile handled by this wave
    const int arow  = lane & 15;
    const int kbase = (lane >> 4) << 4;        // 0 or 16 (A fragment K-half)

    v8f acc = {};
    #pragma unroll
    for (int s = 0; s < 4; ++s) {              // K = 128 in 4 steps of 32
        union { v16bf v; uint4 q[2]; } af, bfr;
        const __bf16* ap = &As[arow * 136 + s * 32 + kbase];
        af.q[0] = *(const uint4*)ap;           // K = s*32 + kbase + 0..7
        af.q[1] = *(const uint4*)(ap + 8);     // K = s*32 + kbase + 8..15
        const __bf16* bp = Bp + ((((nt << 2) + s) << 5) + lane) * 16;
        bfr.q[0] = *(const uint4*)bp;
        bfr.q[1] = *(const uint4*)(bp + 8);
        acc = __builtin_amdgcn_wmma_f32_16x16x32_bf16(false, af.v, false, bfr.v,
                                                      (short)0, acc, false, false);
    }

    const int col   = (nt << 4) + (lane & 15);
    const int rbase = row0 + ((lane >> 4) << 3);
    #pragma unroll
    for (int r = 0; r < 8; ++r)
        C[(rbase + r) * ncols + col] = acc[r];
}

// ---------------- host launcher ----------------

extern "C" void kernel_launch(void* const* d_in, const int* in_sizes, int n_in,
                              void* d_out, int out_size, void* d_ws, size_t ws_size,
                              hipStream_t stream) {
    const float* features = (const float*)d_in[0];
    const int*   src      = (const int*)d_in[1];
    const int*   dst      = (const int*)d_in[2];
    const float* W1       = (const float*)d_in[3];
    const float* b1       = (const float*)d_in[4];
    const float* W2       = (const float*)d_in[5];
    const float* b2       = (const float*)d_in[6];
    float* out = (float*)d_out;

    const int N = in_sizes[0] / KDIM;   // 100000 (multiple of 16)
    const int E = in_sizes[1];          // 1600000

    char* ws = (char*)d_ws;
    size_t off = 0;
    __bf16* fb   = (__bf16*)(ws + off); off += (size_t)N * 128 * 2;
    __bf16* hb   = (__bf16*)(ws + off); off += (size_t)N * 128 * 2;
    float*  m    = (float*)(ws + off);  off += (size_t)N * 128 * 4;   // reused for m1 and m2
    float*  agg1 = (float*)(ws + off);  off += (size_t)N * 128 * 4;
    float*  degv = (float*)(ws + off);  off += (size_t)N * 4;
    __bf16* W1p  = (__bf16*)(ws + off); off += 128 * 128 * 2;
    __bf16* W2p  = (__bf16*)(ws + off); off += 128 * 64 * 2;

    // zero accumulators (d_out / d_ws are poisoned by harness)
    hipMemsetAsync(degv, 0, (size_t)N * 4, stream);
    hipMemsetAsync(agg1, 0, (size_t)N * 128 * 4, stream);
    hipMemsetAsync(out,  0, (size_t)N * 64 * 4, stream);

    // prep
    cvt_bf16_kernel<<<(N * 128 + 255) / 256, 256, 0, stream>>>(features, fb, N * 128);
    pack_w_kernel<<<(8 * 4 * 32 * 16 + 255) / 256, 256, 0, stream>>>(W1, W1p, 128);
    pack_w_kernel<<<(4 * 4 * 32 * 16 + 255) / 256, 256, 0, stream>>>(W2, W2p, 64);
    degree_kernel<<<(E + 255) / 256, 256, 0, stream>>>(dst, degv, E);
    deg_inv_kernel<<<(N + 255) / 256, 256, 0, stream>>>(degv, N);

    // layer 1: m = fb @ W1 ; agg1 = scatter(m[src] -> dst) ; hb = relu(agg1*dinv + b1)
    gemm_bf16_wmma_kernel<<<N / 16, 256, 0, stream>>>(fb, W1p, m, 128);
    long long t1 = (long long)E * 32;
    scatter_add_kernel<<<(int)((t1 + 255) / 256), 256, 0, stream>>>(m, src, dst, agg1, E, 5, 128);
    relu_norm_kernel<<<(N * 128 + 255) / 256, 256, 0, stream>>>(agg1, degv, b1, hb, N * 128);

    // layer 2: m = hb @ W2 ; out = scatter(m[src] -> dst) ; out = out*dinv + b2
    gemm_bf16_wmma_kernel<<<N / 16, 128, 0, stream>>>(hb, W2p, m, 64);
    long long t2 = (long long)E * 16;
    scatter_add_kernel<<<(int)((t2 + 255) / 256), 256, 0, stream>>>(m, src, dst, out, E, 4, 64);
    finalize_kernel<<<(N * 64 + 255) / 256, 256, 0, stream>>>(out, degv, b2, N * 64);
}
